// MiniUNet_52682068853170
// MI455X (gfx1250) — matrix operand order repaired and checked by reading, not compiled, not run on hardware
//
#include <hip/hip_runtime.h>

typedef __attribute__((ext_vector_type(8)))  float  v8f;
typedef __attribute__((ext_vector_type(8)))  __bf16 v8bf;
typedef __attribute__((ext_vector_type(16))) __bf16 v16bf;

__device__ __forceinline__ unsigned short f2bf(float f) {
  unsigned u = __float_as_uint(f);
  u += 0x7FFFu + ((u >> 16) & 1u);
  return (unsigned short)(u >> 16);
}
__device__ __forceinline__ float bf2f(unsigned short h) {
  return __uint_as_float(((unsigned)h) << 16);
}

struct ConvDesc {
  int Di, Hi, Wi, ICp;          // input interior dims, channel stride (padded)
  int Do_, Ho, Wo, OCp, ocoff;  // output interior dims, channel stride, chan offset
  int KD, KH, KW;
  int sdl, shl, swl;            // log2 strides (all strides are 1 or 2)
  int pd, ph, pw;               // conv: pad ; tconv: pl = k-1-fwd_pad
  int nkc;                      // ceil(IC/32) K-chunks
  int mode;                     // 0 = conv, 1 = transposed conv (gather form)
  int relu;
  int hasres;
};

// Implicit-GEMM conv via V_WMMA_F32_16X16X32_BF16.
// Wave tile: 16 output sites (M) x NO*16 output channels (N) -- A reused NO times.
// A: activations channel-last bf16 with 1-voxel zero halo.
// B: weights prepacked into exact wave32 WMMA B layout (contiguous per lane).
template <int NO>
__global__ __launch_bounds__(128) void conv_wmma(
    const unsigned short* __restrict__ in, unsigned short* __restrict__ out,
    const unsigned short* __restrict__ wpk, const float* __restrict__ scale,
    const float* __restrict__ shift, const float* __restrict__ mask,
    const unsigned short* __restrict__ res, ConvDesc d)
{
  const int lane = threadIdx.x & 31;
  const int wv   = threadIdx.x >> 5;
  const int nsites = d.Do_ * d.Ho * d.Wo;
  const int Mbase = (blockIdx.x * 4 + wv) * 16;
  if (Mbase >= nsites) return;              // wave-uniform: EXEC stays all-ones
  const int octile0 = blockIdx.y * NO;
  const int noct    = gridDim.y * NO;

  // per-lane A row (site), unsigned divmod (cheap expansion)
  int s0 = Mbase + (lane & 15);
  if (s0 >= nsites) s0 = nsites - 1;        // clamp: loads valid, stores guarded
  unsigned us0 = (unsigned)s0;
  unsigned uWo = (unsigned)d.Wo, uHo = (unsigned)d.Ho;
  int ow = (int)(us0 % uWo); unsigned t0 = us0 / uWo;
  int oh = (int)(t0 % uHo); int od = (int)(t0 / uHo);

  const int W2  = d.Wi + 2;
  const int HW2 = (d.Hi + 2) * (d.Wi + 2);
  const int kco = (lane >> 4) << 3;         // 0 or 8: A K-offset per lane half
  const int taps = d.KD * d.KH * d.KW;
  const int sdm = (1 << d.sdl) - 1, shm = (1 << d.shl) - 1, swm = (1 << d.swl) - 1;

  v8f acc[NO];
#pragma unroll
  for (int j = 0; j < NO; ++j) { v8f z = {}; acc[j] = z; }

  for (int tap = 0; tap < taps; ++tap) {
    int kw = tap % d.KW; int t1 = tap / d.KW;
    int kh = t1 % d.KH;  int kd = t1 / d.KH;
    int id, ih, iw; bool ok = true;
    if (d.mode == 0) {
      id = (od << d.sdl) + kd - d.pd;       // halo guarantees in-bounds
      ih = (oh << d.shl) + kh - d.ph;
      iw = (ow << d.swl) + kw - d.pw;
    } else {
      int td = od + kd - d.pd, th = oh + kh - d.ph, tw = ow + kw - d.pw;
      ok = (td >= 0) && (th >= 0) && (tw >= 0) &&
           ((td & sdm) == 0) && ((th & shm) == 0) && ((tw & swm) == 0);
      id = td >> d.sdl; ih = th >> d.shl; iw = tw >> d.swl;
      ok = ok && (id < d.Di) && (ih < d.Hi) && (iw < d.Wi);
      if (!ok) { id = 0; ih = 0; iw = 0; }
    }
    size_t hsite = (size_t)(id + 1) * HW2 + (size_t)(ih + 1) * W2 + (iw + 1);
    const unsigned short* ip0 = in + hsite * (size_t)d.ICp + kco;
    if (tap + 1 < taps)                     // hint next tap's weight block into cache
      __builtin_prefetch(wpk + (((size_t)(tap + 1) * d.nkc) * noct + octile0) * 512, 0, 3);
    for (int kc = 0; kc < d.nkc; ++kc) {
      v8bf alo = *(const v8bf*)(ip0 + kc * 32);
      v8bf ahi = *(const v8bf*)(ip0 + kc * 32 + 16);
      if (!ok) { v8bf z = {}; alo = z; ahi = z; }
      v16bf A = __builtin_shufflevector(alo, ahi, 0,1,2,3,4,5,6,7,8,9,10,11,12,13,14,15);
      const unsigned short* bbase =
          wpk + ((((size_t)tap * d.nkc + kc) * noct) + octile0) * 512 + lane * 16;
#pragma unroll
      for (int j = 0; j < NO; ++j) {
        v8bf blo = *(const v8bf*)(bbase + j * 512);
        v8bf bhi = *(const v8bf*)(bbase + j * 512 + 8);
        v16bf B = __builtin_shufflevector(blo, bhi, 0,1,2,3,4,5,6,7,8,9,10,11,12,13,14,15);
        acc[j] = __builtin_amdgcn_wmma_f32_16x16x32_bf16(
            /*neg_a=*/false, A, /*neg_b=*/false, B,
            /*c_mod=*/(short)0, acc[j], /*reuse_a=*/false, /*reuse_b=*/false);
      }
    }
  }

  // epilogue: affine (+residual) (+relu) * mask, store bf16
  float sc[NO], sb[NO];
  int ocl[NO];
#pragma unroll
  for (int j = 0; j < NO; ++j) {
    int oc = (octile0 + j) * 16 + (lane & 15);
    ocl[j] = oc; sc[j] = scale[oc]; sb[j] = shift[oc];
  }
  const int Wo2  = d.Wo + 2;
  const int HWo2 = (d.Ho + 2) * (d.Wo + 2);
  int so_base = Mbase + ((lane >> 4) << 3);
  unsigned usb = (unsigned)(so_base < nsites ? so_base : 0);
  unsigned ow2 = usb % uWo; unsigned q = usb / uWo;
  unsigned oh2 = q % uHo;   unsigned od2 = q / uHo;
  for (int r = 0; r < 8; ++r) {
    int so = so_base + r;
    if (so < nsites) {
      size_t hso = (size_t)(od2 + 1) * HWo2 + (size_t)(oh2 + 1) * Wo2 + (ow2 + 1);
      float mval = mask[so];
#pragma unroll
      for (int j = 0; j < NO; ++j) {
        size_t oidx = hso * (size_t)d.OCp + (size_t)(d.ocoff + ocl[j]);
        float v = acc[j][r] * sc[j] + sb[j];
        if (d.hasres) v += bf2f(res[oidx]);
        if (d.relu && v < 0.f) v = 0.f;
        v *= mval;
        out[oidx] = f2bf(v);
      }
    }
    // carry-increment (od2, oh2, ow2) instead of divmod per row
    if (++ow2 == uWo) { ow2 = 0; if (++oh2 == uHo) { oh2 = 0; ++od2; } }
  }
}

// Pack f32 weights (O,I,KD,KH,KW) into WMMA B layout:
// [tap][kchunk][octile][lane(32)][elem(16)] bf16, lane%16 = N (oc), K = (lane/16)*8+(e&7)+(e/8)*16.
__global__ void prepack_w(const float* __restrict__ w, unsigned short* __restrict__ dst,
                          int O, int I, int taps, int nkc, int noct, int total)
{
  int idx = blockIdx.x * blockDim.x + threadIdx.x;
  if (idx >= total) return;
  unsigned u = (unsigned)idx;
  int e    = u & 15;
  int lane = (u >> 4) & 31;
  unsigned rem = u >> 9;
  int octile = (int)(rem % (unsigned)noct); rem /= (unsigned)noct;
  int kc   = (int)(rem % (unsigned)nkc);
  int tap  = (int)(rem / (unsigned)nkc);
  int K  = ((lane >> 4) << 3) + (e & 7) + ((e >> 3) << 4);
  int ic = kc * 32 + K;
  int oc = octile * 16 + (lane & 15);
  float v = 0.f;
  if (ic < I && oc < O) v = w[((size_t)oc * I + ic) * taps + tap];
  dst[idx] = f2bf(v);
}

__global__ void dilate_mask(const float* __restrict__ in, float* __restrict__ out,
                            int Di, int Hi, int Wi, int Do_, int Ho, int Wo,
                            int KD, int KH, int KW, int sd, int sh, int sw,
                            int pd, int ph, int pw)
{
  int idx = blockIdx.x * blockDim.x + threadIdx.x;
  int n = Do_ * Ho * Wo;
  if (idx >= n) return;
  unsigned u = (unsigned)idx;
  int ow = (int)(u % (unsigned)Wo); unsigned t = u / (unsigned)Wo;
  int oh = (int)(t % (unsigned)Ho); int od = (int)(t / (unsigned)Ho);
  float any = 0.f;
  for (int kd = 0; kd < KD; ++kd) {
    int id = od * sd + kd - pd; if (id < 0 || id >= Di) continue;
    for (int kh = 0; kh < KH; ++kh) {
      int ih = oh * sh + kh - ph; if (ih < 0 || ih >= Hi) continue;
      for (int kw = 0; kw < KW; ++kw) {
        int iw = ow * sw + kw - pw; if (iw < 0 || iw >= Wi) continue;
        if (in[((size_t)id * Hi + ih) * Wi + iw] > 0.f) any = 1.f;
      }
    }
  }
  out[idx] = any;
}

__global__ void prep_input(const float* __restrict__ x, const int* __restrict__ am,
                           unsigned short* __restrict__ dst, float* __restrict__ m1,
                           int D, int H, int W, int C, int Cp)
{
  int idx = blockIdx.x * blockDim.x + threadIdx.x;
  int NS = D * H * W;
  if (idx >= NS) return;
  float m = am[idx] > 0 ? 1.f : 0.f;
  m1[idx] = m;
  unsigned u = (unsigned)idx;
  int w = (int)(u % (unsigned)W); unsigned t = u / (unsigned)W;
  int h = (int)(t % (unsigned)H); int d = (int)(t / (unsigned)H);
  size_t hs = ((size_t)(d + 1) * (H + 2) + (h + 1)) * (W + 2) + (w + 1);
  for (int c = 0; c < C; ++c)
    dst[hs * Cp + c] = f2bf(x[(size_t)c * NS + idx] * m);
}

__global__ void max_over_z(const float* __restrict__ in, float* __restrict__ out,
                           int Z, int HW)
{
  int idx = blockIdx.x * blockDim.x + threadIdx.x;
  if (idx >= HW) return;
  float v = 0.f;
  for (int z = 0; z < Z; ++z) v = fmaxf(v, in[(size_t)z * HW + idx]);
  out[idx] = v;
}

// x5 [Z,H,W,Cp5 halo] -> bev [1,H,W,Cpb halo], bev channel = c*Z + z
__global__ void bev_pack(const unsigned short* __restrict__ x5,
                         unsigned short* __restrict__ bev,
                         int Z, int H, int W, int C, int Cp5, int Cpb)
{
  int idx = blockIdx.x * blockDim.x + threadIdx.x;
  int total = H * W * Z * C;
  if (idx >= total) return;
  unsigned u = (unsigned)idx;
  int c = (int)(u % (unsigned)C); unsigned t = u / (unsigned)C;
  int z = (int)(t % (unsigned)Z); unsigned t2 = t / (unsigned)Z;
  int x = (int)(t2 % (unsigned)W); int y = (int)(t2 / (unsigned)W);
  size_t src = (((size_t)(z + 1) * (H + 2) + (y + 1)) * (W + 2) + (x + 1)) * Cp5 + c;
  size_t dst = (((size_t)1 * (H + 2) + (y + 1)) * (W + 2) + (x + 1)) * Cpb + (c * Z + z);
  bev[dst] = x5[src];
}

// x5 = (x5 + bevout[y,x,:]) * m5
__global__ void scatter_bev(unsigned short* __restrict__ x5,
                            const unsigned short* __restrict__ bevout,
                            const float* __restrict__ m5,
                            int Z, int H, int W, int C, int Cp)
{
  int idx = blockIdx.x * blockDim.x + threadIdx.x;
  int total = Z * H * W * C;
  if (idx >= total) return;
  unsigned u = (unsigned)idx;
  int c = (int)(u % (unsigned)C); unsigned t = u / (unsigned)C;
  int x = (int)(t % (unsigned)W); t /= (unsigned)W;
  int y = (int)(t % (unsigned)H); int z = (int)(t / (unsigned)H);
  size_t hi = (((size_t)(z + 1) * (H + 2) + (y + 1)) * (W + 2) + (x + 1)) * Cp + c;
  size_t bi = (((size_t)1 * (H + 2) + (y + 1)) * (W + 2) + (x + 1)) * Cp + c;
  float v = (bf2f(x5[hi]) + bf2f(bevout[bi])) * m5[((size_t)z * H + y) * W + x];
  x5[hi] = f2bf(v);
}

// halo bf16 channel-last -> NCDHW f32
__global__ void final_out(const unsigned short* __restrict__ src, float* __restrict__ dst,
                          int D, int H, int W, int C, int Cp)
{
  int idx = blockIdx.x * blockDim.x + threadIdx.x;
  int NS = D * H * W;
  int total = NS * C;
  if (idx >= total) return;
  unsigned u = (unsigned)idx;
  int s = (int)(u % (unsigned)NS); int c = (int)(u / (unsigned)NS);
  unsigned us = (unsigned)s;
  int w = (int)(us % (unsigned)W); unsigned t = us / (unsigned)W;
  int h = (int)(t % (unsigned)H); int d = (int)(t / (unsigned)H);
  size_t hi = (((size_t)(d + 1) * (H + 2) + (h + 1)) * (W + 2) + (w + 1)) * Cp + c;
  dst[idx] = bf2f(src[hi]);
}

// ---------------- host side ----------------

struct LayerDef { int O, I, kd, kh, kw; };
static const LayerDef g_layers[38] = {
  {16,5,3,3,3},                                                  // 0  in
  {16,16,3,3,3},{16,16,3,3,3},{16,16,3,3,3},{16,16,3,3,3},       // 1-4  c1_*
  {32,16,3,3,3},                                                 // 5  c2_d
  {32,32,3,3,3},{32,32,3,3,3},{32,32,3,3,3},{32,32,3,3,3},       // 6-9  c2_*
  {64,32,3,3,3},                                                 // 10 c3_d
  {64,64,3,3,3},{64,64,3,3,3},{64,64,3,3,3},{64,64,3,3,3},       // 11-14 c3_*
  {128,64,3,3,3},                                                // 15 c4_d
  {128,128,3,3,3},{128,128,3,3,3},{128,128,3,3,3},{128,128,3,3,3}, // 16-19 c4_*
  {128,128,3,1,1},                                               // 20 c5_d
  {128,256,1,3,3},                                               // 21 m1_d
  {128,128,1,3,3},{128,128,1,3,3},{128,128,1,3,3},               // 22-24 m1_s*
  {128,128,1,3,3},                                               // 25 m2_d
  {128,128,1,3,3},{128,128,1,3,3},{128,128,1,3,3},               // 26-28 m2_s*
  {128,128,1,3,3},                                               // 29 m_inv1
  {128,128,1,3,3},                                               // 30 m_inv2
  {128,256,1,3,3},                                               // 31 m_out
  {128,128,3,3,3},                                               // 32 t5
  {128,128,3,1,1},                                               // 33 inv5
  {128,256,3,3,3},                                               // 34 m5
  {64,64,3,3,3},                                                 // 35 t4
  {64,128,3,3,3},                                                // 36 inv4
  {64,128,3,3,3},                                                // 37 m4
};

extern "C" void kernel_launch(void* const* d_in, const int* in_sizes, int n_in,
                              void* d_out, int out_size, void* d_ws, size_t ws_size,
                              hipStream_t stream)
{
  (void)in_sizes; (void)n_in; (void)out_size;
  const float* voxel = (const float*)d_in[0];
  const int*   amask = (const int*)d_in[1];
  auto PW = [&](int l) { return (const float*)d_in[2 + 3 * l + 0]; };
  auto PS = [&](int l) { return (const float*)d_in[2 + 3 * l + 1]; };
  auto PB = [&](int l) { return (const float*)d_in[2 + 3 * l + 2]; };

  char* ws = (char*)d_ws;
  size_t off = 0;
  auto alloc = [&](size_t bytes) -> char* {
    char* p = ws + off; off += (bytes + 255) & ~(size_t)255; return p;
  };
  auto allocH = [&](int D, int H, int W, int C) -> unsigned short* {
    return (unsigned short*)alloc((size_t)(D + 2) * (H + 2) * (W + 2) * C * 2);
  };
  auto allocF = [&](size_t n) -> float* { return (float*)alloc(n * 4); };

  const int D1 = 41, H1 = 96, W1 = 96;
  const int D2 = 21, H2 = 48, W2 = 48;
  const int D3 = 11, H3 = 24, W3 = 24;
  const int D4 = 5,  H4 = 12, W4 = 12;
  const int D5 = 2,  H5 = 12, W5 = 12;
  const int HB_ = 12, WB_ = 12, HG = 6, WG = 6;

  unsigned short *A1 = allocH(D1, H1, W1, 32),  *Bu1 = allocH(D1, H1, W1, 32);
  unsigned short *A2 = allocH(D2, H2, W2, 32),  *Bu2 = allocH(D2, H2, W2, 32);
  unsigned short *A3 = allocH(D3, H3, W3, 64),  *Bu3 = allocH(D3, H3, W3, 64);
  unsigned short *A4 = allocH(D4, H4, W4, 128), *Bu4 = allocH(D4, H4, W4, 128);
  unsigned short *X5    = allocH(D5, H5, W5, 128);
  unsigned short *BEVIN = allocH(1, HB_, WB_, 256);
  unsigned short *HA    = allocH(1, HB_, WB_, 128), *HBuf = allocH(1, HB_, WB_, 128);
  unsigned short *GA    = allocH(1, HG, WG, 128),   *GB   = allocH(1, HG, WG, 128);
  unsigned short *CATB  = allocH(1, HB_, WB_, 256);
  unsigned short *BEVO  = allocH(1, HB_, WB_, 128);
  unsigned short *CAT5  = allocH(D4, H4, W4, 256);
  unsigned short *CAT4  = allocH(D3, H3, W3, 128);
  unsigned short *OUTB  = allocH(D3, H3, W3, 64);

  float *m1  = allocF((size_t)D1 * H1 * W1);
  float *m2  = allocF((size_t)D2 * H2 * W2);
  float *m3  = allocF((size_t)D3 * H3 * W3);
  float *m4m = allocF((size_t)D4 * H4 * W4);
  float *m5m = allocF((size_t)D5 * H5 * W5);
  float *mbv = allocF((size_t)HB_ * WB_);
  float *mA  = allocF((size_t)HB_ * WB_);
  float *mB  = allocF((size_t)HG * WG);

  unsigned short* wpk[38];
  for (int l = 0; l < 38; ++l) {
    const LayerDef& L = g_layers[l];
    int taps = L.kd * L.kh * L.kw, nkc = (L.I + 31) / 32, noct = L.O / 16;
    wpk[l] = (unsigned short*)alloc((size_t)taps * nkc * noct * 512 * 2);
  }

  size_t used = off;
  hipMemsetAsync(d_ws, 0, used > ws_size ? ws_size : used, stream);

  for (int l = 0; l < 38; ++l) {
    const LayerDef& L = g_layers[l];
    int taps = L.kd * L.kh * L.kw, nkc = (L.I + 31) / 32, noct = L.O / 16;
    int total = taps * nkc * noct * 512;
    prepack_w<<<(total + 255) / 256, 256, 0, stream>>>(PW(l), wpk[l], L.O, L.I, taps, nkc, noct, total);
  }

  {
    int NS = D1 * H1 * W1;
    prep_input<<<(NS + 255) / 256, 256, 0, stream>>>(voxel, amask, A1, m1, D1, H1, W1, 5, 32);
  }

  auto conv = [&](const unsigned short* in, unsigned short* out, int l,
                  const float* mask, const unsigned short* res,
                  int Di, int Hi, int Wi, int ICp,
                  int Do, int Ho, int Wo, int OCp, int ocoff,
                  int sd, int sh, int sw, int pd, int ph, int pw,
                  int mode, int relu) {
    const LayerDef& L = g_layers[l];
    ConvDesc d;
    d.Di = Di; d.Hi = Hi; d.Wi = Wi; d.ICp = ICp;
    d.Do_ = Do; d.Ho = Ho; d.Wo = Wo; d.OCp = OCp; d.ocoff = ocoff;
    d.KD = L.kd; d.KH = L.kh; d.KW = L.kw;
    d.sdl = sd >> 1; d.shl = sh >> 1; d.swl = sw >> 1;   // strides are 1 or 2
    d.pd = pd; d.ph = ph; d.pw = pw;
    d.nkc = (L.I + 31) / 32; d.mode = mode; d.relu = relu;
    d.hasres = (res != nullptr);
    int nsites = Do * Ho * Wo;
    int noct = L.O / 16;
    int NO = noct >= 4 ? 4 : noct;
    dim3 g((nsites + 63) / 64, noct / NO);
    const unsigned short* rp = res ? res : out;
    if (NO == 4)
      conv_wmma<4><<<g, 128, 0, stream>>>(in, out, wpk[l], PS(l), PB(l), mask, rp, d);
    else if (NO == 2)
      conv_wmma<2><<<g, 128, 0, stream>>>(in, out, wpk[l], PS(l), PB(l), mask, rp, d);
    else
      conv_wmma<1><<<g, 128, 0, stream>>>(in, out, wpk[l], PS(l), PB(l), mask, rp, d);
  };

  auto dil = [&](const float* in, float* out, int Di, int Hi, int Wi,
                 int Do, int Ho, int Wo, int KD, int KH, int KW,
                 int sd, int sh, int sw, int pd, int ph, int pw) {
    int n = Do * Ho * Wo;
    dilate_mask<<<(n + 255) / 256, 256, 0, stream>>>(in, out, Di, Hi, Wi, Do, Ho, Wo,
                                                     KD, KH, KW, sd, sh, sw, pd, ph, pw);
  };

  // ---- encoder ----
  conv(A1, Bu1, 0, m1, nullptr, D1,H1,W1,32, D1,H1,W1,32,0, 1,1,1, 1,1,1, 0,1);   // in
  conv(Bu1, A1, 1, m1, nullptr, D1,H1,W1,32, D1,H1,W1,32,0, 1,1,1, 1,1,1, 0,1);   // c1_1a
  conv(A1, Bu1, 2, m1, Bu1,     D1,H1,W1,32, D1,H1,W1,32,0, 1,1,1, 1,1,1, 0,1);   // c1_1b
  conv(Bu1, A1, 3, m1, nullptr, D1,H1,W1,32, D1,H1,W1,32,0, 1,1,1, 1,1,1, 0,1);   // c1_2a
  conv(A1, Bu1, 4, m1, Bu1,     D1,H1,W1,32, D1,H1,W1,32,0, 1,1,1, 1,1,1, 0,1);   // c1_2b

  dil(m1, m2, D1,H1,W1, D2,H2,W2, 3,3,3, 2,2,2, 1,1,1);
  conv(Bu1, A2, 5, m2, nullptr, D1,H1,W1,32, D2,H2,W2,32,0, 2,2,2, 1,1,1, 0,1);   // c2_d
  conv(A2, Bu2, 6, m2, nullptr, D2,H2,W2,32, D2,H2,W2,32,0, 1,1,1, 1,1,1, 0,1);
  conv(Bu2, A2, 7, m2, A2,      D2,H2,W2,32, D2,H2,W2,32,0, 1,1,1, 1,1,1, 0,1);
  conv(A2, Bu2, 8, m2, nullptr, D2,H2,W2,32, D2,H2,W2,32,0, 1,1,1, 1,1,1, 0,1);
  conv(Bu2, A2, 9, m2, A2,      D2,H2,W2,32, D2,H2,W2,32,0, 1,1,1, 1,1,1, 0,1);

  dil(m2, m3, D2,H2,W2, D3,H3,W3, 3,3,3, 2,2,2, 1,1,1);
  conv(A2, A3, 10, m3, nullptr, D2,H2,W2,32, D3,H3,W3,64,0, 2,2,2, 1,1,1, 0,1);   // c3_d
  conv(A3, Bu3, 11, m3, nullptr, D3,H3,W3,64, D3,H3,W3,64,0, 1,1,1, 1,1,1, 0,1);
  conv(Bu3, A3, 12, m3, A3,      D3,H3,W3,64, D3,H3,W3,64,0, 1,1,1, 1,1,1, 0,1);
  conv(A3, Bu3, 13, m3, nullptr, D3,H3,W3,64, D3,H3,W3,64,0, 1,1,1, 1,1,1, 0,1);
  conv(Bu3, A3, 14, m3, A3,      D3,H3,W3,64, D3,H3,W3,64,0, 1,1,1, 1,1,1, 0,1);

  dil(m3, m4m, D3,H3,W3, D4,H4,W4, 3,3,3, 2,2,2, 0,1,1);
  conv(A3, A4, 15, m4m, nullptr, D3,H3,W3,64, D4,H4,W4,128,0, 2,2,2, 0,1,1, 0,1); // c4_d
  conv(A4, Bu4, 16, m4m, nullptr, D4,H4,W4,128, D4,H4,W4,128,0, 1,1,1, 1,1,1, 0,1);
  conv(Bu4, A4, 17, m4m, A4,      D4,H4,W4,128, D4,H4,W4,128,0, 1,1,1, 1,1,1, 0,1);
  conv(A4, Bu4, 18, m4m, nullptr, D4,H4,W4,128, D4,H4,W4,128,0, 1,1,1, 1,1,1, 0,1);
  conv(Bu4, A4, 19, m4m, A4,      D4,H4,W4,128, D4,H4,W4,128,0, 1,1,1, 1,1,1, 0,1);

  dil(m4m, m5m, D4,H4,W4, D5,H5,W5, 3,1,1, 2,1,1, 0,0,0);
  conv(A4, X5, 20, m5m, nullptr, D4,H4,W4,128, D5,H5,W5,128,0, 2,1,1, 0,0,0, 0,1); // c5_d

  // ---- BEV middle ----
  max_over_z<<<1, 256, 0, stream>>>(m5m, mbv, D5, H5 * W5);
  dil(mbv, mA, 1,HB_,WB_, 1,HB_,WB_, 1,3,3, 1,1,1, 0,1,1);
  {
    int tot = HB_ * WB_ * D5 * 128;
    bev_pack<<<(tot + 255) / 256, 256, 0, stream>>>(X5, BEVIN, D5, HB_, WB_, 128, 128, 256);
  }
  conv(BEVIN, HA, 21, mA, nullptr, 1,HB_,WB_,256, 1,HB_,WB_,128,0, 1,1,1, 0,1,1, 0,1); // m1_d
  conv(HA, HBuf, 22, mA, nullptr, 1,HB_,WB_,128, 1,HB_,WB_,128,0, 1,1,1, 0,1,1, 0,1);
  conv(HBuf, HA, 23, mA, nullptr, 1,HB_,WB_,128, 1,HB_,WB_,128,0, 1,1,1, 0,1,1, 0,1);
  conv(HA, HBuf, 24, mA, nullptr, 1,HB_,WB_,128, 1,HB_,WB_,128,0, 1,1,1, 0,1,1, 0,1);
  dil(mA, mB, 1,HB_,WB_, 1,HG,WG, 1,3,3, 1,2,2, 0,1,1);
  conv(HBuf, GA, 25, mB, nullptr, 1,HB_,WB_,128, 1,HG,WG,128,0, 1,2,2, 0,1,1, 0,1); // m2_d
  conv(GA, GB, 26, mB, nullptr, 1,HG,WG,128, 1,HG,WG,128,0, 1,1,1, 0,1,1, 0,1);
  conv(GB, GA, 27, mB, nullptr, 1,HG,WG,128, 1,HG,WG,128,0, 1,1,1, 0,1,1, 0,1);
  conv(GA, GB, 28, mB, nullptr, 1,HG,WG,128, 1,HG,WG,128,0, 1,1,1, 0,1,1, 0,1);
  conv(HBuf, CATB, 29, mA, nullptr, 1,HB_,WB_,128, 1,HB_,WB_,256,0,   1,1,1, 0,1,1, 0,1); // m_inv1
  conv(GB,   CATB, 30, mA, nullptr, 1,HG,WG,128,   1,HB_,WB_,256,128, 1,2,2, 0,1,1, 1,1); // m_inv2 (tconv)
  conv(CATB, BEVO, 31, mA, nullptr, 1,HB_,WB_,256, 1,HB_,WB_,128,0,   1,1,1, 0,1,1, 0,1); // m_out
  {
    int tot = D5 * H5 * W5 * 128;
    scatter_bev<<<(tot + 255) / 256, 256, 0, stream>>>(X5, BEVO, m5m, D5, H5, W5, 128, 128);
  }

  // ---- decoder ----
  conv(A4, CAT5, 32, m4m, nullptr, D4,H4,W4,128, D4,H4,W4,256,0,   1,1,1, 1,1,1, 0,1); // t5
  conv(X5, CAT5, 33, m4m, nullptr, D5,H5,W5,128, D4,H4,W4,256,128, 2,1,1, 2,0,0, 1,1); // inv5 (tconv)
  conv(CAT5, Bu4, 34, m4m, nullptr, D4,H4,W4,256, D4,H4,W4,128,0,  1,1,1, 1,1,1, 0,1); // m5
  conv(A3, CAT4, 35, m3, nullptr, D3,H3,W3,64,  D3,H3,W3,128,0,    1,1,1, 1,1,1, 0,1); // t4
  conv(Bu4, CAT4, 36, m3, nullptr, D4,H4,W4,128, D3,H3,W3,128,64,  2,2,2, 2,1,1, 1,1); // inv4 (tconv)
  conv(CAT4, OUTB, 37, m3, nullptr, D3,H3,W3,128, D3,H3,W3,64,0,   1,1,1, 1,1,1, 0,1); // m4
  {
    int tot = 64 * D3 * H3 * W3;
    final_out<<<(tot + 255) / 256, 256, 0, stream>>>(OUTB, (float*)d_out, D3, H3, W3, 64, 64);
  }
}
